// UncertaintyEstimatorRecursiveNarrowMambaBlock_68719477065
// MI455X (gfx1250) — matrix operand order repaired and checked by reading, not compiled, not run on hardware
//
#include <hip/hip_runtime.h>
#include <hip/hip_bf16.h>

// ---------------------------------------------------------------------------
// UncertaintyEstimatorRecursiveNarrowMambaBlock for MI455X (gfx1250, wave32)
// GEMM-shaped work -> v_wmma_f32_16x16x32_f16 (f16 in, f32 acc).
//  - Each wave computes a 16xN strip (up to 4 accumulators) so one A fragment
//    feeds 4 back-to-back WMMAs (XDL co-exec friendly).
//  - A fragments: two contiguous global_load_b128 per lane (layout-exact).
//  - B fragments: weights pre-swizzled fragment-ready -> 2x b128 per lane.
//  - OOB conv rows read a zeroed scratch row (no divergent per-element loads).
// Selective scan: chunked parallel prefix (64 chunks of 64 steps) since each
// state is a diagonal linear recurrence h = a_t*h + b_t. 64x more parallelism
// than the serial scan at 2x the exp work; fully deterministic.
// ---------------------------------------------------------------------------

typedef _Float16 h16;
typedef __attribute__((ext_vector_type(16))) _Float16 v16h;
typedef __attribute__((ext_vector_type(8)))  _Float16 v8h;
typedef __attribute__((ext_vector_type(8)))  float    v8f;

#define HW      4096            // 64*64
#define NPIX    (6*HW)          // BL * HW = 24576
#define EPSV    1e-5f

__device__ __forceinline__ float silu_f(float x){ return x / (1.f + __expf(-x)); }
__device__ __forceinline__ float gelu_f(float x){
  float x3 = x*x*x;
  return 0.5f*x*(1.f + tanhf(0.7978845608028654f*(x + 0.044715f*x3)));
}
__device__ __forceinline__ int scan_pixel(int k, int t){
  if      (k == 0) return t;
  else if (k == 1) return ((t & 63) << 6) | (t >> 6);
  else if (k == 2) return 4095 - t;
  else { int tt = 4095 - t; return ((tt & 63) << 6) | (tt >> 6); }
}

// ---- WMMA fragment builders (CDNA5 16-bit A/B layouts, wave32) -------------
__device__ __forceinline__ v16h load_a_row(const h16* __restrict__ arow, int kbase, int lane){
  int half = lane >> 4;
  v8h lo = *(const v8h*)(arow + kbase + 8*half);
  v8h hi = *(const v8h*)(arow + kbase + 16 + 8*half);
  return __builtin_shufflevector(lo, hi, 0,1,2,3,4,5,6,7,8,9,10,11,12,13,14,15);
}
__device__ __forceinline__ v16h load_b_packed(const h16* __restrict__ wp, int kstep,
                                              int NT, int ntile, int lane){
  const h16* p = wp + (((size_t)kstep*NT + ntile)*32 + lane)*16;
  v8h lo = *(const v8h*)(p);
  v8h hi = *(const v8h*)(p + 8);
  return __builtin_shufflevector(lo, hi, 0,1,2,3,4,5,6,7,8,9,10,11,12,13,14,15);
}
#define WMMA_F16(acc, a, b) \
  acc = __builtin_amdgcn_wmma_f32_16x16x32_f16(false, (a), false, (b), (short)0, (acc), false, false)

// ---------------------------------------------------------------------------
// Reductions / normalization
// ---------------------------------------------------------------------------
__global__ void k_meanvar(const float* __restrict__ x, float* __restrict__ stats, int n){
  int g = blockIdx.x;
  const float* p = x + (size_t)g * n;
  float s = 0.f, ss = 0.f;
  for (int i = threadIdx.x; i < n; i += 256){ float v = p[i]; s += v; ss += v*v; }
  __shared__ float sh[512];
  sh[threadIdx.x] = s; sh[256 + threadIdx.x] = ss; __syncthreads();
  for (int o = 128; o > 0; o >>= 1){
    if (threadIdx.x < o){ sh[threadIdx.x] += sh[threadIdx.x+o]; sh[256+threadIdx.x] += sh[256+threadIdx.x+o]; }
    __syncthreads();
  }
  if (threadIdx.x == 0){
    float m = sh[0] / n; float v = sh[256] / n - m*m;
    stats[g*2] = m; stats[g*2+1] = rsqrtf(v + EPSV);
  }
}

__global__ void k_build_x0(const float* __restrict__ img, const float* __restrict__ dz,
                           const float* __restrict__ sg, const float* __restrict__ stats,
                           float* __restrict__ x0){
  int i  = blockIdx.x*256 + threadIdx.x;      // < NPIX*192
  int ch = i % 192; int t = i / 192; int hw = t & (HW-1); int bl = t >> 12;
  int b  = bl / 3;
  float v;
  if (ch < 64)        v = dz[((size_t)(b*64 + ch))*HW + hw];
  else if (ch < 128){ float m = stats[bl*2], rs = stats[bl*2+1];
                      v = (img[((size_t)(bl*64 + (ch-64)))*HW + hw] - m)*rs; }
  else                v = sg[((size_t)(b*64 + (ch-128)))*HW + hw];
  x0[i] = v;
}

__global__ void k_gn_silu1(const float* __restrict__ x0, const float* __restrict__ stats,
                           const float* __restrict__ g, const float* __restrict__ b,
                           h16* __restrict__ a0h, h16* __restrict__ x0h){
  int i  = blockIdx.x*256 + threadIdx.x;      // < NPIX*192
  int ch = i % 192; int bl = i / (192*HW);
  float m = stats[bl*2], rs = stats[bl*2+1];
  float v = x0[i];
  a0h[i] = (h16)silu_f((v - m)*rs*g[ch] + b[ch]);
  x0h[i] = (h16)v;
}

__global__ void k_gn_silu2(const float* __restrict__ h1, const float* __restrict__ stats,
                           const float* __restrict__ g, const float* __restrict__ b,
                           h16* __restrict__ h1h){
  int i  = blockIdx.x*256 + threadIdx.x;      // < NPIX*64
  int ch = i & 63; int bl = i >> 18;
  float m = stats[bl*2], rs = stats[bl*2+1];
  h1h[i] = (h16)silu_f((h1[i] - m)*rs*g[ch] + b[ch]);
}

__global__ void k_ln64(const float* __restrict__ x, const float* __restrict__ g,
                       const float* __restrict__ b, h16* __restrict__ out){
  __shared__ float sh[64];
  int row = blockIdx.x, c = threadIdx.x;
  float v = x[(size_t)row*64 + c];
  sh[c] = v; __syncthreads();
  for (int o = 32; o > 0; o >>= 1){ if (c < o) sh[c] += sh[c+o]; __syncthreads(); }
  float m = sh[0] * (1.f/64.f); __syncthreads();
  float d = v - m; sh[c] = d*d; __syncthreads();
  for (int o = 32; o > 0; o >>= 1){ if (c < o) sh[c] += sh[c+o]; __syncthreads(); }
  float rs = rsqrtf(sh[0]*(1.f/64.f) + EPSV);
  out[(size_t)row*64 + c] = (h16)(d*rs*g[c] + b[c]);
}

__global__ void k_outgate(const float* __restrict__ ysum, const float* __restrict__ xz,
                          const float* __restrict__ g, const float* __restrict__ b,
                          h16* __restrict__ out){
  __shared__ float sh[128];
  int row = blockIdx.x, c = threadIdx.x;
  float v = ysum[(size_t)row*128 + c];
  sh[c] = v; __syncthreads();
  for (int o = 64; o > 0; o >>= 1){ if (c < o) sh[c] += sh[c+o]; __syncthreads(); }
  float m = sh[0]*(1.f/128.f); __syncthreads();
  float d = v - m; sh[c] = d*d; __syncthreads();
  for (int o = 64; o > 0; o >>= 1){ if (c < o) sh[c] += sh[c+o]; __syncthreads(); }
  float rs = rsqrtf(sh[0]*(1.f/128.f) + EPSV);
  float z  = xz[(size_t)row*256 + 128 + c];
  out[(size_t)row*128 + c] = (h16)((d*rs*g[c] + b[c]) * silu_f(z));
}

// ---------------------------------------------------------------------------
// Weight repack kernels (f32 -> f16 [K][N], then fragment-ready packing)
// ---------------------------------------------------------------------------
__global__ void k_wT(const float* __restrict__ w, h16* __restrict__ o, int N, int Kd){
  int i = blockIdx.x*256 + threadIdx.x;
  if (i >= N*Kd) return;
  int n = i / Kd, k2 = i % Kd;
  o[(size_t)k2*N + n] = (h16)w[i];
}
__global__ void k_wconv(const float* __restrict__ w, h16* __restrict__ o, int cin){
  int i = blockIdx.x*256 + threadIdx.x;
  if (i >= 64*cin*9) return;
  int co = i/(cin*9); int rem = i%(cin*9); int ci = rem/9; int tap = rem%9;
  o[((size_t)tap*cin + ci)*64 + co] = (h16)w[i];
}
__global__ void k_wxp(const float* __restrict__ w, h16* __restrict__ o){
  int i = blockIdx.x*256 + threadIdx.x;       // 4*36*128
  if (i >= 4*36*128) return;
  int k = i/(36*128); int rem = i%(36*128); int c = rem/128; int d = rem%128;
  o[((size_t)(k*128 + d))*36 + c] = (h16)w[i];
}
__global__ void k_prepA(const float* __restrict__ a_logs, float* __restrict__ An){
  int i = blockIdx.x*256 + threadIdx.x;       // 4*128*16 = 8192
  if (i >= 8192) return;
  An[i] = -__expf(a_logs[i]);
}
__global__ void k_packB(const h16* __restrict__ w, h16* __restrict__ o,
                        int K, int N, int NT){
  int i = blockIdx.x*256 + threadIdx.x;
  int KS = (K + 31) >> 5;
  if (i >= KS*NT*512) return;
  int e = i & 15; int lane = (i >> 4) & 31; int nt = (i >> 9) % NT; int ks = i / (512*NT);
  int k = ks*32 + 8*(lane >> 4) + (e & 7) + 16*(e >> 3);
  int n = nt*16 + (lane & 15);
  o[i] = (k < K && n < N) ? w[(size_t)k*N + n] : (h16)0.f;
}
__global__ void k_zero_h16(h16* __restrict__ p, int n){
  int i = blockIdx.x*256 + threadIdx.x;
  if (i < n) p[i] = (h16)0.f;
}

// ---------------------------------------------------------------------------
// WMMA implicit-GEMM kernels (exact grids; no partial waves; EXEC all ones)
// ---------------------------------------------------------------------------
template<int CIN>
__global__ void k_conv3x3(const h16* __restrict__ act, const h16* __restrict__ wp,
                          const float* __restrict__ bias, float* __restrict__ dst,
                          const h16* __restrict__ zrow, int accumulate){
  int gt = blockIdx.x*256 + threadIdx.x;
  int wave = gt >> 5, lane = gt & 31;
  int mtile = wave;                           // [0,1536)
  int m = mtile*16 + (lane & 15);
  int bl = m >> 12, hw = m & (HW-1), y = hw >> 6, x = hw & 63;
  v8f acc[4] = {};
  #pragma unroll
  for (int tap = 0; tap < 9; ++tap){
    int yy = y + tap/3 - 1, xx = x + tap%3 - 1;
    bool ok = (yy >= 0 && yy < 64 && xx >= 0 && xx < 64);
    const h16* arow = ok ? act + ((size_t)((bl << 12) + (yy << 6) + xx))*CIN : zrow;
    #pragma unroll
    for (int kb = 0; kb < CIN; kb += 32){
      int kstep = (tap*CIN + kb) >> 5;
      v16h av = load_a_row(arow, kb, lane);
      #pragma unroll
      for (int nt = 0; nt < 4; ++nt)
        WMMA_F16(acc[nt], av, load_b_packed(wp, kstep, 4, nt, lane));
    }
  }
  int half = lane >> 4;
  #pragma unroll
  for (int nt = 0; nt < 4; ++nt){
    int col = nt*16 + (lane & 15);
    float bc = bias[col];
    #pragma unroll
    for (int j = 0; j < 8; ++j){
      int row = mtile*16 + j + 8*half;
      size_t idx = (size_t)row*64 + col;
      float v = acc[nt][j] + bc;
      if (accumulate) dst[idx] += v; else dst[idx] = v;
    }
  }
}

__global__ void k_skip(const h16* __restrict__ a, const h16* __restrict__ wp,
                       const float* __restrict__ bias, float* __restrict__ dst){
  int gt = blockIdx.x*256 + threadIdx.x;
  int wave = gt >> 5, lane = gt & 31;
  int mtile = wave;
  const h16* arow = a + (size_t)(mtile*16 + (lane & 15))*192;
  v8f acc[4] = {};
  #pragma unroll
  for (int kb = 0; kb < 192; kb += 32){
    v16h av = load_a_row(arow, kb, lane);
    #pragma unroll
    for (int nt = 0; nt < 4; ++nt)
      WMMA_F16(acc[nt], av, load_b_packed(wp, kb >> 5, 4, nt, lane));
  }
  int half = lane >> 4;
  #pragma unroll
  for (int nt = 0; nt < 4; ++nt){
    int col = nt*16 + (lane & 15);
    float bc = bias[col];
    #pragma unroll
    for (int j = 0; j < 8; ++j){
      int row = mtile*16 + j + 8*half;
      dst[(size_t)row*64 + col] = acc[nt][j] + bc;
    }
  }
}

__global__ void k_inproj(const h16* __restrict__ a, const h16* __restrict__ wp,
                         float* __restrict__ xz){
  int gt = blockIdx.x*256 + threadIdx.x;
  int wave = gt >> 5, lane = gt & 31;
  int ngrp = wave & 3, mtile = wave >> 2;
  const h16* arow = a + (size_t)(mtile*16 + (lane & 15))*64;
  v8f acc[4] = {};
  #pragma unroll
  for (int kb = 0; kb < 64; kb += 32){
    v16h av = load_a_row(arow, kb, lane);
    #pragma unroll
    for (int nt = 0; nt < 4; ++nt)
      WMMA_F16(acc[nt], av, load_b_packed(wp, kb >> 5, 16, ngrp*4 + nt, lane));
  }
  int half = lane >> 4;
  #pragma unroll
  for (int nt = 0; nt < 4; ++nt){
    int col = (ngrp*4 + nt)*16 + (lane & 15);
    #pragma unroll
    for (int j = 0; j < 8; ++j){
      int row = mtile*16 + j + 8*half;
      xz[(size_t)row*256 + col] = acc[nt][j];
    }
  }
}

__global__ void k_xproj(const h16* __restrict__ xah, const h16* __restrict__ wxpp,
                        float* __restrict__ xdbl){
  int gt = blockIdx.x*256 + threadIdx.x;
  int wave = gt >> 5, lane = gt & 31;
  int mt = wave & 255; int k = (wave >> 8) & 3; int bl = wave >> 10;
  int t = mt*16 + (lane & 15);
  int p = scan_pixel(k, t);
  const h16* arow = xah + ((size_t)((bl << 12) + p))*128;
  const h16* wb   = wxpp + (size_t)k*(4*3*512);   // KS=4, NT=3
  v8f acc[3] = {};
  #pragma unroll
  for (int kb = 0; kb < 128; kb += 32){
    v16h av = load_a_row(arow, kb, lane);
    #pragma unroll
    for (int nt = 0; nt < 3; ++nt)
      WMMA_F16(acc[nt], av, load_b_packed(wb, kb >> 5, 3, nt, lane));
  }
  int half = lane >> 4;
  #pragma unroll
  for (int nt = 0; nt < 3; ++nt){
    int col = nt*16 + (lane & 15);
    if (col < 36){
      #pragma unroll
      for (int j = 0; j < 8; ++j){
        int tr = mt*16 + j + 8*half;
        xdbl[((size_t)((bl*4 + k)*4096 + tr))*36 + col] = acc[nt][j];
      }
    }
  }
}

__global__ void k_outproj(const h16* __restrict__ a, const h16* __restrict__ wp,
                          const float* __restrict__ xres, float* __restrict__ x2){
  int gt = blockIdx.x*256 + threadIdx.x;
  int wave = gt >> 5, lane = gt & 31;
  int mtile = wave;
  const h16* arow = a + (size_t)(mtile*16 + (lane & 15))*128;
  v8f acc[4] = {};
  #pragma unroll
  for (int kb = 0; kb < 128; kb += 32){
    v16h av = load_a_row(arow, kb, lane);
    #pragma unroll
    for (int nt = 0; nt < 4; ++nt)
      WMMA_F16(acc[nt], av, load_b_packed(wp, kb >> 5, 4, nt, lane));
  }
  int half = lane >> 4;
  #pragma unroll
  for (int nt = 0; nt < 4; ++nt){
    int col = nt*16 + (lane & 15);
    #pragma unroll
    for (int j = 0; j < 8; ++j){
      int row = mtile*16 + j + 8*half;
      size_t idx = (size_t)row*64 + col;
      x2[idx] = xres[idx] + acc[nt][j];
    }
  }
}

__global__ void k_fc1(const h16* __restrict__ a, const h16* __restrict__ wp,
                      const float* __restrict__ bias, h16* __restrict__ gh){
  int gt = blockIdx.x*256 + threadIdx.x;
  int wave = gt >> 5, lane = gt & 31;
  int mtile = wave;
  const h16* arow = a + (size_t)(mtile*16 + (lane & 15))*64;
  v8f acc[4] = {};
  #pragma unroll
  for (int kb = 0; kb < 64; kb += 32){
    v16h av = load_a_row(arow, kb, lane);
    #pragma unroll
    for (int nt = 0; nt < 4; ++nt)
      WMMA_F16(acc[nt], av, load_b_packed(wp, kb >> 5, 4, nt, lane));
  }
  int half = lane >> 4;
  #pragma unroll
  for (int nt = 0; nt < 4; ++nt){
    int col = nt*16 + (lane & 15);
    float bc = bias[col];
    #pragma unroll
    for (int j = 0; j < 8; ++j){
      int row = mtile*16 + j + 8*half;
      gh[(size_t)row*64 + col] = (h16)gelu_f(acc[nt][j] + bc);
    }
  }
}

__global__ void k_fc2(const h16* __restrict__ a, const h16* __restrict__ wp,
                      const float* __restrict__ bias, const float* __restrict__ x2,
                      float* __restrict__ out){
  int gt = blockIdx.x*256 + threadIdx.x;
  int wave = gt >> 5, lane = gt & 31;
  int mtile = wave;
  const h16* arow = a + (size_t)(mtile*16 + (lane & 15))*64;
  v8f acc[4] = {};
  #pragma unroll
  for (int kb = 0; kb < 64; kb += 32){
    v16h av = load_a_row(arow, kb, lane);
    #pragma unroll
    for (int nt = 0; nt < 4; ++nt)
      WMMA_F16(acc[nt], av, load_b_packed(wp, kb >> 5, 4, nt, lane));
  }
  int half = lane >> 4;
  #pragma unroll
  for (int nt = 0; nt < 4; ++nt){
    int col = nt*16 + (lane & 15);
    float bc = bias[col];
    #pragma unroll
    for (int j = 0; j < 8; ++j){
      int row = mtile*16 + j + 8*half;
      float v = x2[(size_t)row*64 + col] + acc[nt][j] + bc;
      int bl = row >> 12, hw = row & (HW-1);
      out[(size_t)bl*64*HW + (size_t)col*HW + hw] = v;
    }
  }
}

// ---------------------------------------------------------------------------
// Depthwise conv + SiLU, dt_proj + softplus
// ---------------------------------------------------------------------------
__global__ void k_dwconv(const float* __restrict__ xz, const float* __restrict__ w,
                         const float* __restrict__ bias, float* __restrict__ xa,
                         h16* __restrict__ xah){
  int i = blockIdx.x*256 + threadIdx.x;       // < NPIX*128
  int d = i & 127; int t = i >> 7; int hw = t & (HW-1); int bl = t >> 12;
  int y = hw >> 6, x = hw & 63;
  float s = bias[d];
  #pragma unroll
  for (int tap = 0; tap < 9; ++tap){
    int yy = y + tap/3 - 1, xx = x + tap%3 - 1;
    if (yy >= 0 && yy < 64 && xx >= 0 && xx < 64)
      s += w[d*9 + tap] * xz[((size_t)((bl << 12) + (yy << 6) + xx))*256 + d];
  }
  float v = silu_f(s);
  xa[i] = v; xah[i] = (h16)v;
}

__global__ void k_delta(const float* __restrict__ xdbl, const float* __restrict__ dtw,
                        const float* __restrict__ dtb, float* __restrict__ delta){
  int i = blockIdx.x*256 + threadIdx.x;       // < 24*4096*128
  int d = i & 127; int t = (i >> 7) & 4095; int bk = i >> 19; int k = bk & 3;
  const float* xr = xdbl + ((size_t)(bk*4096 + t))*36;
  const float* wr = dtw + (size_t)(k*128 + d)*4;
  float s = dtb[k*128 + d];
  #pragma unroll
  for (int r = 0; r < 4; ++r) s += xr[r]*wr[r];
  delta[i] = (s > 20.f) ? s : log1pf(__expf(s));
}

// ---------------------------------------------------------------------------
// Chunked parallel selective scan: 64 chunks x 64 steps.
// Phase 1: per chunk, Aprod = prod(dA), Bacc = h_end with h0=0.
// Phase 2: serial prefix over chunks -> h_start per chunk.
// Phase 3: re-run chunk from h_start, emit y.
// Layout: Aprod/Bacc/hstart [bk][chunk][d][16] (16 contiguous per thread).
// ---------------------------------------------------------------------------
__global__ void k_scan_p1(const float* __restrict__ delta, const float* __restrict__ xdbl,
                          const float* __restrict__ xa, const float* __restrict__ An,
                          float* __restrict__ Aprod, float* __restrict__ Bacc){
  __shared__ float sB[16];
  int blkid = blockIdx.x; int bk = blkid >> 6; int c = blkid & 63;
  int bl = bk >> 2; int k = bk & 3; int d = threadIdx.x;
  float A[16];
  #pragma unroll
  for (int n = 0; n < 16; ++n) A[n] = An[(size_t)(k*128 + d)*16 + n];
  float ap[16], hb[16];
  #pragma unroll
  for (int n = 0; n < 16; ++n){ ap[n] = 1.f; hb[n] = 0.f; }
  const float* dl_base = delta + (size_t)bk*4096*128;
  const float* xd_base = xdbl  + (size_t)bk*4096*36;
  const float* xa_bl   = xa    + (size_t)bl*4096*128;
  int t0 = c*64;
  for (int tt = 0; tt < 64; ++tt){
    int t = t0 + tt;
    if (d < 16) sB[d] = xd_base[(size_t)t*36 + 4 + d];
    __syncthreads();
    float u  = xa_bl[(size_t)scan_pixel(k, t)*128 + d];
    float dl = dl_base[(size_t)t*128 + d];
    #pragma unroll
    for (int n = 0; n < 16; ++n){
      float dA = __expf(dl * A[n]);
      ap[n] *= dA;
      hb[n] = dA*hb[n] + dl*sB[n]*u;
    }
    __syncthreads();
  }
  size_t base = (((size_t)bk*64 + c)*128 + d)*16;
  #pragma unroll
  for (int n = 0; n < 16; ++n){ Aprod[base + n] = ap[n]; Bacc[base + n] = hb[n]; }
}

__global__ void k_scan_p2(const float* __restrict__ Aprod, const float* __restrict__ Bacc,
                          float* __restrict__ hstart){
  int i = blockIdx.x*256 + threadIdx.x;       // < 24*128*16
  int n = i & 15; int d = (i >> 4) & 127; int bk = i >> 11;
  float hs = 0.f;
  for (int c = 0; c < 64; ++c){
    size_t idx = (((size_t)bk*64 + c)*128 + d)*16 + n;
    hstart[idx] = hs;
    hs = Aprod[idx]*hs + Bacc[idx];
  }
}

__global__ void k_scan_p3(const float* __restrict__ delta, const float* __restrict__ xdbl,
                          const float* __restrict__ xa, const float* __restrict__ An,
                          const float* __restrict__ Ds, const float* __restrict__ hstart,
                          float* __restrict__ oy){
  __shared__ float sB[16], sC[16];
  int blkid = blockIdx.x; int bk = blkid >> 6; int c = blkid & 63;
  int bl = bk >> 2; int k = bk & 3; int d = threadIdx.x;
  float A[16];
  #pragma unroll
  for (int n = 0; n < 16; ++n) A[n] = An[(size_t)(k*128 + d)*16 + n];
  float Dv = Ds[k*128 + d];
  float h[16];
  size_t hbase = (((size_t)bk*64 + c)*128 + d)*16;
  #pragma unroll
  for (int n = 0; n < 16; ++n) h[n] = hstart[hbase + n];
  const float* dl_base = delta + (size_t)bk*4096*128;
  const float* xd_base = xdbl  + (size_t)bk*4096*36;
  const float* xa_bl   = xa    + (size_t)bl*4096*128;
  float*       oy_base = oy    + (size_t)bk*4096*128;
  int t0 = c*64;
  for (int tt = 0; tt < 64; ++tt){
    int t = t0 + tt;
    if (d < 32){
      float v = xd_base[(size_t)t*36 + 4 + d];
      if (d < 16) sB[d] = v; else sC[d - 16] = v;
    }
    __syncthreads();
    float u  = xa_bl[(size_t)scan_pixel(k, t)*128 + d];
    float dl = dl_base[(size_t)t*128 + d];
    float y = 0.f;
    #pragma unroll
    for (int n = 0; n < 16; ++n){
      float dA = __expf(dl * A[n]);
      h[n] = dA*h[n] + dl*sB[n]*u;
      y += h[n]*sC[n];
    }
    oy_base[(size_t)t*128 + d] = y + u*Dv;
    __syncthreads();
  }
}

__global__ void k_combine(const float* __restrict__ oy, float* __restrict__ ysum){
  int i = blockIdx.x*256 + threadIdx.x;       // < NPIX*128
  int d = i & 127; int p = (i >> 7) & 4095; int bl = i >> 19;
  int y0 = p >> 6, x0 = p & 63;
  int t1 = (x0 << 6) | y0;
  const float* base = oy + (size_t)bl*4*4096*128;
  float v = base[(size_t)(0*4096 + p)        *128 + d]
          + base[(size_t)(1*4096 + t1)       *128 + d]
          + base[(size_t)(2*4096 + (4095-p)) *128 + d]
          + base[(size_t)(3*4096 + (4095-t1))*128 + d];
  ysum[i] = v;
}

// ---------------------------------------------------------------------------
// Host launcher
// ---------------------------------------------------------------------------
extern "C" void kernel_launch(void* const* d_in, const int* in_sizes, int n_in,
                              void* d_out, int out_size, void* d_ws, size_t ws_size,
                              hipStream_t stream) {
  const float* img      = (const float*)d_in[0];
  const float* dz       = (const float*)d_in[1];
  const float* sg       = (const float*)d_in[2];
  const float* gn1_g    = (const float*)d_in[3];
  const float* gn1_b    = (const float*)d_in[4];
  const float* conv1_w  = (const float*)d_in[5];
  const float* conv1_b  = (const float*)d_in[6];
  const float* gn2_g    = (const float*)d_in[7];
  const float* gn2_b    = (const float*)d_in[8];
  const float* conv2_w  = (const float*)d_in[9];
  const float* conv2_b  = (const float*)d_in[10];
  const float* skip_w   = (const float*)d_in[11];
  const float* skip_b   = (const float*)d_in[12];
  const float* ln1_g    = (const float*)d_in[13];
  const float* ln1_b    = (const float*)d_in[14];
  const float* ln2_g    = (const float*)d_in[15];
  const float* ln2_b    = (const float*)d_in[16];
  const float* in_proj_w= (const float*)d_in[17];
  const float* dwconv_w = (const float*)d_in[18];
  const float* dwconv_b = (const float*)d_in[19];
  const float* x_proj_w = (const float*)d_in[20];
  const float* dt_proj_w= (const float*)d_in[21];
  const float* dt_proj_b= (const float*)d_in[22];
  const float* A_logs   = (const float*)d_in[23];
  const float* Ds_in    = (const float*)d_in[24];
  const float* onorm_g  = (const float*)d_in[25];
  const float* onorm_b  = (const float*)d_in[26];
  const float* out_proj_w=(const float*)d_in[27];
  const float* fc1_w    = (const float*)d_in[28];
  const float* fc1_b    = (const float*)d_in[29];
  const float* fc2_w    = (const float*)d_in[30];
  const float* fc2_b    = (const float*)d_in[31];
  float* out = (float*)d_out;

  char* ws = (char*)d_ws;
  size_t off = 0;
  auto alloc = [&](size_t bytes) -> char* {
    char* p = ws + off; off += (bytes + 255) & ~(size_t)255; return p;
  };
  float* stats_img = (float*)alloc(48);
  float* stats_g1  = (float*)alloc(48);
  float* stats_g2  = (float*)alloc(48);
  float* x0   = (float*)alloc((size_t)NPIX*192*4);
  h16*   x0h  = (h16*)  alloc((size_t)NPIX*192*2);
  h16*   a0h  = (h16*)  alloc((size_t)NPIX*192*2);
  float* h1   = (float*)alloc((size_t)NPIX*64*4);
  h16*   h1h  = (h16*)  alloc((size_t)NPIX*64*2);
  float* xres = (float*)alloc((size_t)NPIX*64*4);
  h16*   xlnh = (h16*)  alloc((size_t)NPIX*64*2);
  float* xz   = (float*)alloc((size_t)NPIX*256*4);
  float* xa   = (float*)alloc((size_t)NPIX*128*4);
  h16*   xah  = (h16*)  alloc((size_t)NPIX*128*2);
  float* xdbl = (float*)alloc((size_t)24*4096*36*4);
  float* delta= (float*)alloc((size_t)24*4096*128*4);
  float* oy   = (float*)alloc((size_t)24*4096*128*4);
  float* Aprod= (float*)alloc((size_t)24*64*128*16*4);
  float* Bacc = (float*)alloc((size_t)24*64*128*16*4);
  float* hst  = (float*)alloc((size_t)24*64*128*16*4);
  float* ysum = (float*)alloc((size_t)NPIX*128*4);
  h16*   ygh  = (h16*)  alloc((size_t)NPIX*128*2);
  float* x2   = (float*)alloc((size_t)NPIX*64*4);
  h16*   x2lnh= (h16*)  alloc((size_t)NPIX*64*2);
  h16*   gh   = (h16*)  alloc((size_t)NPIX*64*2);
  // staging [K][N] f16 weights
  h16*   w_in = (h16*)  alloc((size_t)64*256*2);
  h16*   w1   = (h16*)  alloc((size_t)1728*64*2);
  h16*   w2   = (h16*)  alloc((size_t)576*64*2);
  h16*   wskip= (h16*)  alloc((size_t)192*64*2);
  h16*   wxp  = (h16*)  alloc((size_t)4*128*36*2);
  h16*   wout = (h16*)  alloc((size_t)128*64*2);
  h16*   wfc1 = (h16*)  alloc((size_t)64*64*2);
  h16*   wfc2 = (h16*)  alloc((size_t)64*64*2);
  // fragment-ready packed weights
  h16*   w_inp = (h16*)alloc((size_t)2*16*512*2);
  h16*   w1p   = (h16*)alloc((size_t)54*4*512*2);
  h16*   w2p   = (h16*)alloc((size_t)18*4*512*2);
  h16*   wskipp= (h16*)alloc((size_t)6*4*512*2);
  h16*   wxpp  = (h16*)alloc((size_t)4*4*3*512*2);
  h16*   woutp = (h16*)alloc((size_t)4*4*512*2);
  h16*   wfc1p = (h16*)alloc((size_t)2*4*512*2);
  h16*   wfc2p = (h16*)alloc((size_t)2*4*512*2);
  h16*   zrow  = (h16*)alloc((size_t)256*2);
  float* An    = (float*)alloc((size_t)4*128*16*4);
  (void)ws_size; (void)in_sizes; (void)n_in; (void)out_size;

  // ---- weight repacks ([K][N] f16) ----
  k_wT   <<<64, 256, 0, stream>>>(in_proj_w,  w_in, 256, 64);
  k_wT   <<<48, 256, 0, stream>>>(skip_w,     wskip, 64, 192);
  k_wT   <<<32, 256, 0, stream>>>(out_proj_w, wout,  64, 128);
  k_wT   <<<16, 256, 0, stream>>>(fc1_w,      wfc1,  64, 64);
  k_wT   <<<16, 256, 0, stream>>>(fc2_w,      wfc2,  64, 64);
  k_wconv<<<432,256, 0, stream>>>(conv1_w, w1, 192);
  k_wconv<<<144,256, 0, stream>>>(conv2_w, w2, 64);
  k_wxp  <<<72, 256, 0, stream>>>(x_proj_w, wxp);
  k_prepA<<<32, 256, 0, stream>>>(A_logs, An);
  k_zero_h16<<<1, 256, 0, stream>>>(zrow, 256);
  // ---- fragment-ready packing ----
  k_packB<<<64, 256, 0, stream>>>(w_in,  w_inp, 64, 256, 16);
  k_packB<<<432,256, 0, stream>>>(w1,    w1p,   1728, 64, 4);
  k_packB<<<144,256, 0, stream>>>(w2,    w2p,   576,  64, 4);
  k_packB<<<48, 256, 0, stream>>>(wskip, wskipp, 192, 64, 4);
  for (int k = 0; k < 4; ++k)
    k_packB<<<24, 256, 0, stream>>>(wxp + (size_t)k*128*36, wxpp + (size_t)k*(4*3*512),
                                    128, 36, 3);
  k_packB<<<32, 256, 0, stream>>>(wout, woutp, 128, 64, 4);
  k_packB<<<16, 256, 0, stream>>>(wfc1, wfc1p, 64, 64, 4);
  k_packB<<<16, 256, 0, stream>>>(wfc2, wfc2p, 64, 64, 4);

  // ---- frontend: instance LN, concat, GN+SiLU, convs, skip ----
  k_meanvar <<<6, 256, 0, stream>>>(img, stats_img, 64*HW);
  k_build_x0<<<18432, 256, 0, stream>>>(img, dz, sg, stats_img, x0);
  k_meanvar <<<6, 256, 0, stream>>>(x0, stats_g1, 192*HW);
  k_gn_silu1<<<18432, 256, 0, stream>>>(x0, stats_g1, gn1_g, gn1_b, a0h, x0h);
  k_conv3x3<192><<<192, 256, 0, stream>>>(a0h, w1p, conv1_b, h1, zrow, 0);
  k_meanvar <<<6, 256, 0, stream>>>(h1, stats_g2, 64*HW);
  k_gn_silu2<<<6144, 256, 0, stream>>>(h1, stats_g2, gn2_g, gn2_b, h1h);
  k_skip    <<<192, 256, 0, stream>>>(x0h, wskipp, skip_b, xres);
  k_conv3x3<64><<<192, 256, 0, stream>>>(h1h, w2p, conv2_b, xres, zrow, 1);

  // ---- SS2D ----
  k_ln64   <<<NPIX, 64, 0, stream>>>(xres, ln1_g, ln1_b, xlnh);
  k_inproj <<<768, 256, 0, stream>>>(xlnh, w_inp, xz);
  k_dwconv <<<12288, 256, 0, stream>>>(xz, dwconv_w, dwconv_b, xa, xah);
  k_xproj  <<<768, 256, 0, stream>>>(xah, wxpp, xdbl);
  k_delta  <<<49152, 256, 0, stream>>>(xdbl, dt_proj_w, dt_proj_b, delta);
  k_scan_p1<<<1536, 128, 0, stream>>>(delta, xdbl, xa, An, Aprod, Bacc);
  k_scan_p2<<<192, 256, 0, stream>>>(Aprod, Bacc, hst);
  k_scan_p3<<<1536, 128, 0, stream>>>(delta, xdbl, xa, An, Ds_in, hst, oy);
  k_combine<<<12288, 256, 0, stream>>>(oy, ysum);
  k_outgate<<<NPIX, 128, 0, stream>>>(ysum, xz, onorm_g, onorm_b, ygh);
  k_outproj<<<192, 256, 0, stream>>>(ygh, woutp, xres, x2);

  // ---- MLP + residual + transpose to NCHW ----
  k_ln64<<<NPIX, 64, 0, stream>>>(x2, ln2_g, ln2_b, x2lnh);
  k_fc1 <<<192, 256, 0, stream>>>(x2lnh, wfc1p, fc1_b, gh);
  k_fc2 <<<192, 256, 0, stream>>>(gh, wfc2p, fc2_b, x2, out);
}
